// MultiModalRegressor_70712341561625
// MI455X (gfx1250) — compile-verified, matrix operand order checked by reading, not run on hardware
//
#include <hip/hip_runtime.h>
#include <hip/hip_bf16.h>

typedef __attribute__((ext_vector_type(16))) _Float16 v16h;
typedef __attribute__((ext_vector_type(8)))  float    v8f;

// ---------------- WMMA fragment index map (CDNA5 16x16x32 f16, wave32) ---------
// A (16x32): lane l holds row l%16; element i holds K = (i<8?i:i+8) + 8*(l>=16)
// B (32x16): lane l holds col l%16; same K map (symmetric layout)
// C/D (16x16 f32, 8 VGPRs): vgpr r -> M = r + 8*(l>=16), N = l%16
__device__ __forceinline__ int kmap32(int lane, int i) {
  return (i < 8 ? i : i + 8) + ((lane & 16) ? 8 : 0);
}

#define WMMA_F16(A_, B_, C_) \
  __builtin_amdgcn_wmma_f32_16x16x32_f16(false, (A_), false, (B_), (short)0, (C_), false, false)

// ---------------- Generic GEMM, 16x16 tile per wave (fallback, small M) --------
// C[M,N] = act(A[M,K] @ B[K,N] + bias[N]); f32 memory, f16 fragments, f32 acc.
template <int ACT>
__global__ void __launch_bounds__(32)
gemm_kernel(const float* __restrict__ A, const float* __restrict__ B,
            const float* __restrict__ bias, float* __restrict__ C,
            int M, int N, int K) {
  const int lane = threadIdx.x;
  const int tn = blockIdx.x << 4;
  const int tm = blockIdx.y << 4;
  const int r16 = lane & 15;
  const int hi  = (lane >> 4) & 1;
  const int row = tm + r16;
  const int col = tn + r16;
  const float* arow = A + (size_t)row * K;
  v8f acc = {};
  for (int k0 = 0; k0 < K; k0 += 32) {
    v16h a, b;
#pragma unroll
    for (int i = 0; i < 16; ++i) {
      const int kk = k0 + kmap32(lane, i);
      a[i] = (_Float16)arow[kk];
      b[i] = (_Float16)B[(size_t)kk * N + col];
    }
    acc = WMMA_F16(a, b, acc);
  }
#pragma unroll
  for (int r = 0; r < 8; ++r) {
    float v = acc[r] + bias[col];
    if (ACT == 1) v = fmaxf(v, 0.0f);
    C[(size_t)(tm + r + (hi << 3)) * N + col] = v;
  }
}

// ---------------- Generic GEMM, 32x32 register-tiled (2x2 WMMA per k-step) -----
// Each wave computes a 32x32 output block: A loaded once per 2 column-tiles,
// B loaded once per 2 row-tiles -> 2x arithmetic intensity vs 16x16.
// Requires M%32==0, N%32==0, K%32==0.
template <int ACT>
__global__ void __launch_bounds__(32)
gemm32_kernel(const float* __restrict__ A, const float* __restrict__ B,
              const float* __restrict__ bias, float* __restrict__ C,
              int M, int N, int K) {
  const int lane = threadIdx.x;
  const int tn = blockIdx.x << 5;
  const int tm = blockIdx.y << 5;
  const int r16 = lane & 15;
  const int hi  = (lane >> 4) & 1;
  const int col0 = tn + r16;
  const int col1 = col0 + 16;
  const float* arow0 = A + (size_t)(tm + r16) * K;
  const float* arow1 = A + (size_t)(tm + 16 + r16) * K;
  v8f acc00 = {}, acc01 = {}, acc10 = {}, acc11 = {};
  for (int k0 = 0; k0 < K; k0 += 32) {
    v16h a0, a1, b0, b1;
#pragma unroll
    for (int i = 0; i < 16; ++i) {
      const int kk = k0 + kmap32(lane, i);
      a0[i] = (_Float16)arow0[kk];
      a1[i] = (_Float16)arow1[kk];
      b0[i] = (_Float16)B[(size_t)kk * N + col0];
      b1[i] = (_Float16)B[(size_t)kk * N + col1];
    }
    acc00 = WMMA_F16(a0, b0, acc00);
    acc01 = WMMA_F16(a0, b1, acc01);
    acc10 = WMMA_F16(a1, b0, acc10);
    acc11 = WMMA_F16(a1, b1, acc11);
  }
  const float bia0 = bias[col0], bia1 = bias[col1];
#pragma unroll
  for (int r = 0; r < 8; ++r) {
    const int row0 = tm + r + (hi << 3);
    const int row1 = row0 + 16;
    float v00 = acc00[r] + bia0, v01 = acc01[r] + bia1;
    float v10 = acc10[r] + bia0, v11 = acc11[r] + bia1;
    if (ACT == 1) {
      v00 = fmaxf(v00, 0.0f); v01 = fmaxf(v01, 0.0f);
      v10 = fmaxf(v10, 0.0f); v11 = fmaxf(v11, 0.0f);
    }
    C[(size_t)row0 * N + col0] = v00;
    C[(size_t)row0 * N + col1] = v01;
    C[(size_t)row1 * N + col0] = v10;
    C[(size_t)row1 * N + col1] = v11;
  }
}

// ---------------- EdgeNetwork fused GEMM, 32x32 register-tiled -----------------
// msg[m,e] = sum_{c<49,f<128} ef[m,c] * W2[c*128+f, e] * h[src(m), f]
// A-tiles z[m, c*128+f] = ef[m,c]*h[src(m), f] generated in registers;
// c==48 is the constant-1 channel carrying elin_b exactly. K = 49*128 = 6272.
__global__ void __launch_bounds__(32)
edge_msg_kernel(const float* __restrict__ ef, const float* __restrict__ h,
                const int* __restrict__ eidx, const float* __restrict__ W2,
                float* __restrict__ msg, int M) {
  const int lane = threadIdx.x;
  const int tn = blockIdx.x << 5;      // N = 128
  const int tm = blockIdx.y << 5;
  const int r16 = lane & 15;
  const int hi  = (lane >> 4) & 1;
  const int col0 = tn + r16;
  const int col1 = col0 + 16;
  const int row0 = tm + r16;
  const int row1 = row0 + 16;
  const int src0 = eidx[M + row0];     // edge_index[1][row]
  const int src1 = eidx[M + row1];
  const float* hrow0  = h + (size_t)src0 * 128;
  const float* hrow1  = h + (size_t)src1 * 128;
  const float* efrow0 = ef + (size_t)row0 * 49;
  const float* efrow1 = ef + (size_t)row1 * 49;
  v8f acc00 = {}, acc01 = {}, acc10 = {}, acc11 = {};
  for (int k0 = 0; k0 < 6272; k0 += 32) {
    const float efc0 = efrow0[k0 >> 7]; // 32-wide tile lies in a single c-block
    const float efc1 = efrow1[k0 >> 7];
    const int fb = k0 & 127;
    v16h a0, a1, b0, b1;
#pragma unroll
    for (int i = 0; i < 16; ++i) {
      const int km = kmap32(lane, i);
      a0[i] = (_Float16)(efc0 * hrow0[fb + km]);
      a1[i] = (_Float16)(efc1 * hrow1[fb + km]);
      b0[i] = (_Float16)W2[(size_t)(k0 + km) * 128 + col0];
      b1[i] = (_Float16)W2[(size_t)(k0 + km) * 128 + col1];
    }
    acc00 = WMMA_F16(a0, b0, acc00);
    acc01 = WMMA_F16(a0, b1, acc01);
    acc10 = WMMA_F16(a1, b0, acc10);
    acc11 = WMMA_F16(a1, b1, acc11);
  }
#pragma unroll
  for (int r = 0; r < 8; ++r) {
    const int m0 = tm + r + (hi << 3);
    const int m1 = m0 + 16;
    msg[(size_t)m0 * 128 + col0] = acc00[r];
    msg[(size_t)m0 * 128 + col1] = acc01[r];
    msg[(size_t)m1 * 128 + col0] = acc10[r];
    msg[(size_t)m1 * 128 + col1] = acc11[r];
  }
}

// ---------------- small helper kernels -----------------------------------------
__global__ void build_w2_kernel(const float* __restrict__ elin_w,
                                const float* __restrict__ elin_b,
                                float* __restrict__ W2) {
  int idx = blockIdx.x * 256 + threadIdx.x;   // idx = k*128 + e, K=6272
  if (idx >= 6272 * 128) return;
  int e = idx & 127, k = idx >> 7;
  int c = k >> 7, f = k & 127;
  W2[idx] = (c < 48) ? elin_w[(size_t)c * 16384 + e * 128 + f]
                     : elin_b[e * 128 + f];
}

__global__ void edge_feat_kernel(const float* __restrict__ ea,
                                 float* __restrict__ ef, int M) {
  int m = blockIdx.x, t = threadIdx.x;
  if (t >= 49) return;
  float v;
  if (t < 16) {
    v = ea[m * 17 + t];
  } else if (t < 48) {
    float d = ea[m * 17 + 16];
    d = fminf(fmaxf(d, 0.0f), 10.0f);
    float cen = (float)(t - 16) * (5.0f / 31.0f);
    float x = d - cen;
    v = __expf(-38.44f * x * x);              // gamma = (5/31)^-2
  } else {
    v = 1.0f;                                  // bias channel
  }
  ef[m * 49 + t] = v;
}

__global__ void zero_kernel(float* p, int n) {
  int i = blockIdx.x * 256 + threadIdx.x;
  if (i < n) p[i] = 0.0f;
}

__global__ void scatter_add_kernel(const float* __restrict__ msg,
                                   const int* __restrict__ eidx,
                                   float* __restrict__ aggr, int M) {
  int idx = blockIdx.x * 256 + threadIdx.x;
  if (idx >= M * 128) return;
  int m = idx >> 7, e = idx & 127;
  atomicAdd(&aggr[(size_t)eidx[m] * 128 + e], msg[idx]);
}

// out = LN(a + b)*g + beta over rows of width 128; in-place (out==a) is safe.
__global__ void add_ln_kernel(const float* __restrict__ a, const float* __restrict__ b,
                              const float* __restrict__ g, const float* __restrict__ be,
                              float* __restrict__ out) {
  __shared__ float red[128];
  int r = blockIdx.x, t = threadIdx.x;
  float x = a[(size_t)r * 128 + t] + b[(size_t)r * 128 + t];
  red[t] = x; __syncthreads();
  for (int s = 64; s > 0; s >>= 1) { if (t < s) red[t] += red[t + s]; __syncthreads(); }
  float mu = red[0] / 128.0f; __syncthreads();
  float d = x - mu;
  red[t] = d * d; __syncthreads();
  for (int s = 64; s > 0; s >>= 1) { if (t < s) red[t] += red[t + s]; __syncthreads(); }
  float var = red[0] / 128.0f;
  out[(size_t)r * 128 + t] = d * rsqrtf(var + 1e-5f) * g[t] + be[t];
}

__global__ void pair_dist_kernel(const float* __restrict__ pos, float* __restrict__ dist) {
  int gn = blockIdx.x, m = threadIdx.x;       // gn = g*32+n, m = other node
  int g = gn >> 5, n = gn & 31;
  const float* p = pos + (size_t)g * 96;
  float dx = p[n * 3 + 0] - p[m * 3 + 0];
  float dy = p[n * 3 + 1] - p[m * 3 + 1];
  float dz = p[n * 3 + 2] - p[m * 3 + 2];
  float d2 = dx * dx + dy * dy + dz * dz;
  dist[(size_t)gn * 32 + m] = d2 > 0.0f ? sqrtf(d2) : 0.0f;
}

// graph attention: one block per (graph, head), 32 threads (one per query node)
__global__ void graph_attn_kernel(const float* __restrict__ q, const float* __restrict__ k,
                                  const float* __restrict__ v, const float* __restrict__ dist,
                                  float* __restrict__ o) {
  __shared__ float ks[32 * 16], vs[32 * 16];
  int gh = blockIdx.x, g = gh >> 3, h = gh & 7;
  int n = threadIdx.x;
  for (int d = 0; d < 16; ++d) {
    ks[n * 16 + d] = k[(size_t)(g * 32 + n) * 128 + h * 16 + d];
    vs[n * 16 + d] = v[(size_t)(g * 32 + n) * 128 + h * 16 + d];
  }
  __syncthreads();
  float qv[16];
  for (int d = 0; d < 16; ++d) qv[d] = q[(size_t)(g * 32 + n) * 128 + h * 16 + d];
  float s[32], mx = -1e30f;
  for (int m = 0; m < 32; ++m) {
    float dot = 0.0f;
    for (int d = 0; d < 16; ++d) dot += qv[d] * ks[m * 16 + d];
    float sv = dot * 0.25f + dist[(size_t)(g * 32 + n) * 32 + m];   // 1/sqrt(16)
    sv = fminf(fmaxf(sv, -10.0f), 10.0f);
    s[m] = sv; mx = fmaxf(mx, sv);
  }
  float sum = 0.0f;
  for (int m = 0; m < 32; ++m) { s[m] = __expf(s[m] - mx); sum += s[m]; }
  float inv = 1.0f / sum;
  for (int d = 0; d < 16; ++d) {
    float acc = 0.0f;
    for (int m = 0; m < 32; ++m) acc += s[m] * vs[m * 16 + d];
    o[(size_t)(g * 32 + n) * 128 + h * 16 + d] = acc * inv;
  }
}

// seq attention: one block per (batch, head), 128 threads (one per query pos)
__global__ void seq_attn_kernel(const float* __restrict__ qkv, float* __restrict__ o) {
  __shared__ float ks[128 * 32];
  __shared__ float vs[128 * 32];
  int bh = blockIdx.x, b = bh >> 2, h = bh & 3;
  int n = threadIdx.x;
  const float* base = qkv + (size_t)(b * 128) * 384;
  for (int d = 0; d < 32; ++d) {
    ks[n * 32 + d] = base[(size_t)n * 384 + 128 + h * 32 + d];
    vs[n * 32 + d] = base[(size_t)n * 384 + 256 + h * 32 + d];
  }
  __syncthreads();
  float qv[32];
  for (int d = 0; d < 32; ++d) qv[d] = base[(size_t)n * 384 + h * 32 + d];
  const float scale = 0.17677669529663687f;   // 1/sqrt(32)
  float mx = -1e30f;
  for (int m = 0; m < 128; ++m) {
    float dot = 0.0f;
    for (int d = 0; d < 32; ++d) dot += qv[d] * ks[m * 32 + d];
    mx = fmaxf(mx, dot * scale);
  }
  float sum = 0.0f, acc[32];
  for (int d = 0; d < 32; ++d) acc[d] = 0.0f;
  for (int m = 0; m < 128; ++m) {
    float dot = 0.0f;
    for (int d = 0; d < 32; ++d) dot += qv[d] * ks[m * 32 + d];
    float w = __expf(dot * scale - mx);
    sum += w;
    for (int d = 0; d < 32; ++d) acc[d] += w * vs[m * 32 + d];
  }
  float inv = 1.0f / sum;
  for (int d = 0; d < 32; ++d)
    o[(size_t)(b * 128 + n) * 128 + h * 32 + d] = acc[d] * inv;
}

__global__ void embed_kernel(const int* __restrict__ tokens, const float* __restrict__ tok_emb,
                             const float* __restrict__ pos_emb, float* __restrict__ x) {
  int bl = blockIdx.x, t = threadIdx.x;
  int l = bl & 127;
  int tok = tokens[bl];
  x[(size_t)bl * 128 + t] = tok_emb[(size_t)tok * 128 + t] + pos_emb[(size_t)l * 128 + t];
}

__global__ void graph_pool_kernel(const float* __restrict__ h, float* __restrict__ out) {
  int g = blockIdx.x, e = threadIdx.x;
  float s = 0.0f;
  for (int n = 0; n < 32; ++n) s += h[(size_t)(g * 32 + n) * 128 + e];
  out[(size_t)g * 128 + e] = s / (32.0f + 1e-8f);
}

__global__ void seq_pool_kernel(const float* __restrict__ x, float* __restrict__ out) {
  int b = blockIdx.x, e = threadIdx.x;
  float s = 0.0f;
  for (int l = 0; l < 128; ++l) s += x[(size_t)(b * 128 + l) * 128 + e];
  out[(size_t)b * 128 + e] = s / 128.0f;
}

// total = mean over 4 conformers + scaffold
__global__ void combine_kernel(const float* __restrict__ gconf, const float* __restrict__ gscaf,
                               float* __restrict__ total) {
  int b = blockIdx.x, e = threadIdx.x;
  float s = 0.0f;
  for (int c = 0; c < 4; ++c) s += gconf[(size_t)(b * 4 + c) * 128 + e];
  total[(size_t)b * 128 + e] = s * 0.25f + gscaf[(size_t)b * 128 + e];
}

__global__ void concat_kernel(const float* __restrict__ a, const float* __restrict__ b,
                              float* __restrict__ out) {
  int r = blockIdx.x, t = threadIdx.x;      // 256 threads
  out[(size_t)r * 256 + t] = (t < 128) ? a[(size_t)r * 128 + t]
                                       : b[(size_t)r * 128 + t - 128];
}

__global__ void fuse_kernel(const float* __restrict__ gpre, const float* __restrict__ total,
                            const float* __restrict__ seq, float* __restrict__ fused) {
  int b = blockIdx.x, e = threadIdx.x;
  float g = 1.0f / (1.0f + __expf(-gpre[(size_t)b * 128 + e]));
  fused[(size_t)b * 128 + e] =
      g * total[(size_t)b * 128 + e] + (1.0f - g) * seq[(size_t)b * 128 + e];
}

__global__ void copy2_kernel(const float* __restrict__ src, float* __restrict__ d1,
                             float* __restrict__ d2, int n) {
  int i = blockIdx.x * 256 + threadIdx.x;
  if (i < n) { d1[i] = src[i]; d2[i] = src[i]; }
}

// head1: h[b,e] = relu( sum_k 0.5*(g+s)[b,k]*r1w[k,e] + sum_j gf[b,j]*r1w[128+j,e] + r1b[e] )
__global__ void head1_kernel(const float* __restrict__ g, const float* __restrict__ s,
                             const float* __restrict__ gf, const float* __restrict__ r1w,
                             const float* __restrict__ r1b, float* __restrict__ h) {
  int b = blockIdx.x, e = threadIdx.x;
  float acc = r1b[e];
  for (int k = 0; k < 128; ++k)
    acc += 0.5f * (g[(size_t)b * 128 + k] + s[(size_t)b * 128 + k]) * r1w[(size_t)k * 128 + e];
  for (int j = 0; j < 3; ++j)
    acc += gf[b * 3 + j] * r1w[(size_t)(128 + j) * 128 + e];
  h[(size_t)b * 128 + e] = fmaxf(acc, 0.0f);
}

__global__ void head2_kernel(const float* __restrict__ h, const float* __restrict__ r2w,
                             const float* __restrict__ r2b, float* __restrict__ out) {
  __shared__ float red[128];
  int b = blockIdx.x, t = threadIdx.x;
  red[t] = h[(size_t)b * 128 + t] * r2w[t];
  __syncthreads();
  for (int s = 64; s > 0; s >>= 1) { if (t < s) red[t] += red[t + s]; __syncthreads(); }
  if (t == 0) out[b] = red[0] + r2b[0];
}

// =============================== host orchestration ===========================
extern "C" void kernel_launch(void* const* d_in, const int* in_sizes, int n_in,
                              void* d_out, int out_size, void* d_ws, size_t ws_size,
                              hipStream_t stream) {
  (void)in_sizes; (void)n_in; (void)out_size; (void)ws_size;

  // ---- inputs in setup_inputs() dict order ----
  const float* conf_x   = (const float*)d_in[0];
  const int*   conf_ei  = (const int*)  d_in[1];
  const float* conf_ea  = (const float*)d_in[2];
  const float* conf_pos = (const float*)d_in[3];
  const float* scaf_x   = (const float*)d_in[4];
  const int*   scaf_ei  = (const int*)  d_in[5];
  const float* scaf_ea  = (const float*)d_in[6];
  const float* scaf_pos = (const float*)d_in[7];
  const int*   tokens   = (const int*)  d_in[8];
  const float* gfeat    = (const float*)d_in[9];
  // d_in[10] = conf_per_mol (== 4, matches setup; hardcoded in combine_kernel)
  const float* gproj_w = (const float*)d_in[11];
  const float* gproj_b = (const float*)d_in[12];
  const float* elin_w  = (const float*)d_in[13];
  const float* elin_b  = (const float*)d_in[14];
  const float* eln_g   = (const float*)d_in[15];
  const float* eln_b   = (const float*)d_in[16];
  const float* gt_qw   = (const float*)d_in[17];
  const float* gt_qb   = (const float*)d_in[18];
  const float* gt_kw   = (const float*)d_in[19];
  const float* gt_kb   = (const float*)d_in[20];
  const float* gt_vw   = (const float*)d_in[21];
  const float* gt_vb   = (const float*)d_in[22];
  const float* gt_ow   = (const float*)d_in[23];
  const float* gt_ob   = (const float*)d_in[24];
  const float* gt_ln1g = (const float*)d_in[25];
  const float* gt_ln1b = (const float*)d_in[26];
  const float* gt_f1w  = (const float*)d_in[27];
  const float* gt_f1b  = (const float*)d_in[28];
  const float* gt_f2w  = (const float*)d_in[29];
  const float* gt_f2b  = (const float*)d_in[30];
  const float* gt_ln2g = (const float*)d_in[31];
  const float* gt_ln2b = (const float*)d_in[32];
  const float* tok_emb = (const float*)d_in[33];
  const float* pos_emb = (const float*)d_in[34];
  const float* se_inw  = (const float*)d_in[35];
  const float* se_inb  = (const float*)d_in[36];
  const float* se_ow   = (const float*)d_in[37];
  const float* se_ob   = (const float*)d_in[38];
  const float* se_ln1g = (const float*)d_in[39];
  const float* se_ln1b = (const float*)d_in[40];
  const float* se_f1w  = (const float*)d_in[41];
  const float* se_f1b  = (const float*)d_in[42];
  const float* se_f2w  = (const float*)d_in[43];
  const float* se_f2b  = (const float*)d_in[44];
  const float* se_ln2g = (const float*)d_in[45];
  const float* se_ln2b = (const float*)d_in[46];
  const float* fg1_w   = (const float*)d_in[47];
  const float* fg1_b   = (const float*)d_in[48];
  const float* fg2_w   = (const float*)d_in[49];
  const float* fg2_b   = (const float*)d_in[50];
  const float* cm_g2s_vw = (const float*)d_in[51];
  const float* cm_g2s_vb = (const float*)d_in[52];
  const float* cm_g2s_ow = (const float*)d_in[53];
  const float* cm_g2s_ob = (const float*)d_in[54];
  const float* cm_s2g_vw = (const float*)d_in[55];
  const float* cm_s2g_vb = (const float*)d_in[56];
  const float* cm_s2g_ow = (const float*)d_in[57];
  const float* cm_s2g_ob = (const float*)d_in[58];
  const float* cm_n1g  = (const float*)d_in[59];
  const float* cm_n1b  = (const float*)d_in[60];
  const float* cm_n2g  = (const float*)d_in[61];
  const float* cm_n2b  = (const float*)d_in[62];
  const float* r1_w    = (const float*)d_in[63];
  const float* r1_b    = (const float*)d_in[64];
  const float* r2_w    = (const float*)d_in[65];
  const float* r2_b    = (const float*)d_in[66];

  // ---- workspace bump allocator (floats) ----
  float* ws = (float*)d_ws;
  size_t off = 0;
  auto alloc = [&](size_t n) { float* p = ws + off; off += n; return p; };
  float* W2    = alloc(6272 * 128);   // reordered elin_w + elin_b
  float* ef    = alloc(4096 * 49);
  float* h     = alloc(2048 * 128);
  float* aggr  = alloc(2048 * 128);
  float* qb    = alloc(2048 * 128);
  float* kb    = alloc(2048 * 128);
  float* vb    = alloc(2048 * 128);
  float* ao    = alloc(2048 * 128);
  float* msg   = alloc(4096 * 128);
  float* dist  = alloc(64 * 1024);
  float* ffb   = alloc(2048 * 256);
  float* gconf = alloc(64 * 128);
  float* gscaf = alloc(16 * 128);
  float* sx    = alloc(2048 * 128);
  float* sproj = alloc(2048 * 128);
  float* sqkv  = alloc(2048 * 384);
  float* sao   = alloc(2048 * 128);
  float* sff   = alloc(2048 * 256);
  float* total = alloc(16 * 128);
  float* semb  = alloc(16 * 128);
  float* cat2  = alloc(16 * 256);
  float* ghid  = alloc(16 * 64);
  float* gpre  = alloc(16 * 128);
  float* fus   = alloc(16 * 128);
  float* gB    = alloc(16 * 128);
  float* sB    = alloc(16 * 128);
  float* t1    = alloc(16 * 128);
  float* t2    = alloc(16 * 128);
  float* hh    = alloc(16 * 128);

  // ---- build W2 [6272,128] once (shared by conf & scaf encoders) ----
  {
    int n = 6272 * 128;
    build_w2_kernel<<<(n + 255) / 256, 256, 0, stream>>>(elin_w, elin_b, W2);
  }

  // ---- shared graph encoder ----
  auto run_graph = [&](const float* x, const int* ei, const float* ea,
                       const float* pos, int G, float* gout) {
    const int GN = G * 32, M = G * 64;
    edge_feat_kernel<<<M, 64, 0, stream>>>(ea, ef, M);
    gemm32_kernel<0><<<dim3(4, GN / 32), 32, 0, stream>>>(x, gproj_w, gproj_b, h, GN, 128, 64);
    edge_msg_kernel<<<dim3(4, M / 32), 32, 0, stream>>>(ef, h, ei, W2, msg, M);
    zero_kernel<<<(GN * 128 + 255) / 256, 256, 0, stream>>>(aggr, GN * 128);
    scatter_add_kernel<<<(M * 128 + 255) / 256, 256, 0, stream>>>(msg, ei, aggr, M);
    add_ln_kernel<<<GN, 128, 0, stream>>>(h, aggr, eln_g, eln_b, h);
    pair_dist_kernel<<<GN, 32, 0, stream>>>(pos, dist);
    for (int l = 0; l < 3; ++l) {
      gemm32_kernel<0><<<dim3(4, GN / 32), 32, 0, stream>>>(h, gt_qw + l * 16384, gt_qb + l * 128, qb, GN, 128, 128);
      gemm32_kernel<0><<<dim3(4, GN / 32), 32, 0, stream>>>(h, gt_kw + l * 16384, gt_kb + l * 128, kb, GN, 128, 128);
      gemm32_kernel<0><<<dim3(4, GN / 32), 32, 0, stream>>>(h, gt_vw + l * 16384, gt_vb + l * 128, vb, GN, 128, 128);
      graph_attn_kernel<<<G * 8, 32, 0, stream>>>(qb, kb, vb, dist, ao);
      gemm32_kernel<0><<<dim3(4, GN / 32), 32, 0, stream>>>(ao, gt_ow + l * 16384, gt_ob + l * 128, qb, GN, 128, 128);
      add_ln_kernel<<<GN, 128, 0, stream>>>(h, qb, gt_ln1g + l * 128, gt_ln1b + l * 128, h);   // h = x1
      gemm32_kernel<1><<<dim3(8, GN / 32), 32, 0, stream>>>(h, gt_f1w + l * 32768, gt_f1b + l * 256, ffb, GN, 256, 128);
      gemm32_kernel<0><<<dim3(4, GN / 32), 32, 0, stream>>>(ffb, gt_f2w + l * 32768, gt_f2b + l * 128, qb, GN, 128, 256);
      add_ln_kernel<<<GN, 128, 0, stream>>>(h, qb, gt_ln2g + l * 128, gt_ln2b + l * 128, h);
    }
    graph_pool_kernel<<<G, 128, 0, stream>>>(h, gout);
  };

  run_graph(conf_x, conf_ei, conf_ea, conf_pos, 64, gconf);
  run_graph(scaf_x, scaf_ei, scaf_ea, scaf_pos, 16, gscaf);
  combine_kernel<<<16, 128, 0, stream>>>(gconf, gscaf, total);

  // ---- sequence encoder ----
  embed_kernel<<<2048, 128, 0, stream>>>(tokens, tok_emb, pos_emb, sx);
  for (int l = 0; l < 4; ++l) {
    gemm32_kernel<0><<<dim3(12, 64), 32, 0, stream>>>(sx, se_inw + l * 49152, se_inb + l * 384, sqkv, 2048, 384, 128);
    seq_attn_kernel<<<64, 128, 0, stream>>>(sqkv, sao);
    gemm32_kernel<0><<<dim3(4, 64), 32, 0, stream>>>(sao, se_ow + l * 16384, se_ob + l * 128, sproj, 2048, 128, 128);
    add_ln_kernel<<<2048, 128, 0, stream>>>(sx, sproj, se_ln1g + l * 128, se_ln1b + l * 128, sx);  // sx = x1
    gemm32_kernel<1><<<dim3(8, 64), 32, 0, stream>>>(sx, se_f1w + l * 32768, se_f1b + l * 256, sff, 2048, 256, 128);
    gemm32_kernel<0><<<dim3(4, 64), 32, 0, stream>>>(sff, se_f2w + l * 32768, se_f2b + l * 128, sproj, 2048, 128, 256);
    add_ln_kernel<<<2048, 128, 0, stream>>>(sx, sproj, se_ln2g + l * 128, se_ln2b + l * 128, sx);
  }
  seq_pool_kernel<<<16, 128, 0, stream>>>(sx, semb);

  // ---- fusion gating (M=16 -> 16x16 fallback kernel) ----
  concat_kernel<<<16, 256, 0, stream>>>(total, semb, cat2);
  gemm_kernel<1><<<dim3(4, 1), 32, 0, stream>>>(cat2, fg1_w, fg1_b, ghid, 16, 64, 256);
  gemm_kernel<0><<<dim3(8, 1), 32, 0, stream>>>(ghid, fg2_w, fg2_b, gpre, 16, 128, 64);
  fuse_kernel<<<16, 128, 0, stream>>>(gpre, total, semb, fus);

  // ---- cross-modal attention (length-1 softmax => v_proj then out_proj) ----
  copy2_kernel<<<(2048 + 255) / 256, 256, 0, stream>>>(fus, gB, sB, 2048);
  for (int l = 0; l < 2; ++l) {
    gemm_kernel<0><<<dim3(8, 1), 32, 0, stream>>>(sB, cm_g2s_vw + l * 16384, cm_g2s_vb + l * 128, t1, 16, 128, 128);
    gemm_kernel<0><<<dim3(8, 1), 32, 0, stream>>>(t1, cm_g2s_ow + l * 16384, cm_g2s_ob + l * 128, t2, 16, 128, 128);
    add_ln_kernel<<<16, 128, 0, stream>>>(gB, t2, cm_n1g + l * 128, cm_n1b + l * 128, gB);
    gemm_kernel<0><<<dim3(8, 1), 32, 0, stream>>>(gB, cm_s2g_vw + l * 16384, cm_s2g_vb + l * 128, t1, 16, 128, 128);
    gemm_kernel<0><<<dim3(8, 1), 32, 0, stream>>>(t1, cm_s2g_ow + l * 16384, cm_s2g_ob + l * 128, t2, 16, 128, 128);
    add_ln_kernel<<<16, 128, 0, stream>>>(sB, t2, cm_n2g + l * 128, cm_n2b + l * 128, sB);
  }

  // ---- regression head (K=131 not a multiple of 16 -> VALU) ----
  head1_kernel<<<16, 128, 0, stream>>>(gB, sB, gfeat, r1_w, r1_b, hh);
  head2_kernel<<<16, 128, 0, stream>>>(hh, r2_w, r2_b, (float*)d_out);
}
